// MultiHeadAttention_80187039416418
// MI455X (gfx1250) — matrix operand order repaired and checked by reading, not compiled, not run on hardware
//
#include <hip/hip_runtime.h>
#include <hip/hip_bf16.h>

#define EMB   768
#define HEADS 12
#define HD    64
#define SEQ   4096
#define BATCH 2

typedef __attribute__((ext_vector_type(16))) _Float16 v16h;
typedef __attribute__((ext_vector_type(8)))  float    v8f;

// ---------------------------------------------------------------------------
// WMMA fragment loaders (contiguous k order: element s holds k = 16*(lane>>4) + s, wave32).
// Both loaders produce two contiguous 16B runs per lane -> global_load_b128 x2.
// ---------------------------------------------------------------------------

// A-matrix 16x32 (MxK), 16-bit, row-major storage with given row stride.
// lane m = lane&15; VGPR v holds K = 16*(lane>>4) + 2v + {0,1}
__device__ __forceinline__ v16h load_a_frag(const _Float16* base, int stride) {
  const int lane = threadIdx.x & 31;
  const _Float16* row = base + (size_t)(lane & 15) * stride;
  const int g8 = (lane >> 4) * 16;
  v16h f;
#pragma unroll
  for (int v = 0; v < 8; ++v) {
    const int k = g8 + 2 * v;
    f[2 * v]     = row[k];
    f[2 * v + 1] = row[k + 1];
  }
  return f;
}

// B-matrix 32x16 (KxN), 16-bit, sourced from TRANSPOSED storage (row n holds
// contiguous K).  lane n = lane&15; VGPR v holds K = 16*(lane>>4) + 2v + {0,1}
__device__ __forceinline__ v16h load_b_frag_t(const _Float16* base, int stride) {
  const int lane = threadIdx.x & 31;
  const _Float16* row = base + (size_t)(lane & 15) * stride;
  const int g16 = (lane >> 4) * 16;
  v16h f;
#pragma unroll
  for (int v = 0; v < 8; ++v) {
    const int k = g16 + 2 * v;
    f[2 * v]     = row[k];
    f[2 * v + 1] = row[k + 1];
  }
  return f;
}

__device__ __forceinline__ v8f wmma_f16(v16h a, v16h b, v8f c) {
  return __builtin_amdgcn_wmma_f32_16x16x32_f16(false, a, false, b,
                                                (short)0, c, false, false);
}

// ---------------------------------------------------------------------------
// Conversion kernels
// ---------------------------------------------------------------------------
__global__ void __launch_bounds__(256)
k_cvt(const float* __restrict__ in, _Float16* __restrict__ out, int n) {
  int i = blockIdx.x * 256 + threadIdx.x;
  if (i < n) out[i] = (_Float16)in[i];
}

// W [K=EMB][N=EMB] fp32  ->  Wt [N][K] f16
__global__ void __launch_bounds__(256)
k_cvt_t(const float* __restrict__ in, _Float16* __restrict__ out) {
  int i = blockIdx.x * 256 + threadIdx.x;   // over EMB*EMB
  int n = i / EMB;
  int k = i - n * EMB;
  out[(size_t)n * EMB + k] = (_Float16)in[(size_t)k * EMB + n];
}

// ---------------------------------------------------------------------------
// GEMM: C[M=B*S][N=EMB] = A[M][EMB](f16) * Bt[N][EMB](f16)^T + bias
// One wave -> one 32x64 register tile (2 A-frags x 4 B-frags -> 8 WMMA/K-step;
// 1.5 b128 loads per WMMA instead of 4).
// mode 0: f32 row-major (d_out)
// mode 1: f16 head-split   [B,H,S,HD]   (Q, K)
// mode 2: f16 head-split-T [B,H,HD,S]   (V -> contiguous kv for attention B-frags)
// ---------------------------------------------------------------------------
__global__ void __launch_bounds__(256)
k_gemm(const _Float16* __restrict__ A, const _Float16* __restrict__ Bt,
       const float* __restrict__ bias, _Float16* __restrict__ out_h,
       float* __restrict__ out_f, int mode) {
  const int wid = blockIdx.x * 8 + (threadIdx.x >> 5);
  const int TN = EMB / 64;                      // 12 tiles across N
  const int tm = wid / TN, tn = wid - tm * TN;
  const int m0 = tm * 32, n0 = tn * 64;

  const _Float16* a0 = A  + (size_t)m0 * EMB;
  const _Float16* a1 = a0 + (size_t)16 * EMB;
  const _Float16* b0 = Bt + (size_t)n0 * EMB;

  v8f acc[2][4] = {};
#pragma unroll 2
  for (int kk = 0; kk < EMB; kk += 32) {
    const v16h af0 = load_a_frag(a0 + kk, EMB);
    const v16h af1 = load_a_frag(a1 + kk, EMB);
    const v16h bf0 = load_b_frag_t(b0 + kk,                EMB);
    const v16h bf1 = load_b_frag_t(b0 + 16 * EMB + kk,     EMB);
    const v16h bf2 = load_b_frag_t(b0 + 32 * EMB + kk,     EMB);
    const v16h bf3 = load_b_frag_t(b0 + 48 * EMB + kk,     EMB);
    acc[0][0] = wmma_f16(af0, bf0, acc[0][0]);
    acc[0][1] = wmma_f16(af0, bf1, acc[0][1]);
    acc[0][2] = wmma_f16(af0, bf2, acc[0][2]);
    acc[0][3] = wmma_f16(af0, bf3, acc[0][3]);
    acc[1][0] = wmma_f16(af1, bf0, acc[1][0]);
    acc[1][1] = wmma_f16(af1, bf1, acc[1][1]);
    acc[1][2] = wmma_f16(af1, bf2, acc[1][2]);
    acc[1][3] = wmma_f16(af1, bf3, acc[1][3]);
  }

  const int lane = threadIdx.x & 31;
  const int colL = lane & 15;
  const int rhi  = (lane >> 4) * 8;
#pragma unroll
  for (int mi = 0; mi < 2; ++mi) {
#pragma unroll
    for (int ni = 0; ni < 4; ++ni) {
#pragma unroll
      for (int r = 0; r < 8; ++r) {
        const int row = m0 + mi * 16 + r + rhi;   // 0 .. B*S-1
        const int col = n0 + ni * 16 + colL;      // 0 .. EMB-1
        const float val = acc[mi][ni][r] + bias[col];
        if (mode == 0) {
          out_f[(size_t)row * EMB + col] = val;
        } else {
          const int b = row >> 12;                // / SEQ
          const int s = row & (SEQ - 1);
          const int h = col >> 6;                 // / HD
          const int d = col & (HD - 1);
          if (mode == 1)
            out_h[((((size_t)b * HEADS + h) << 12) + s) * HD + d] = (_Float16)val;
          else
            out_h[(((size_t)b * HEADS + h) * HD + d) * SEQ + s] = (_Float16)val;
        }
      }
    }
  }
}

// ---------------------------------------------------------------------------
// Flash-attention: one wave per 16-row q tile of one (b,h).
// kv blocks of 32; causal; online softmax; fp32 accumulators.
// Q,K in [B,H,S,HD]; V pre-transposed to [B,H,HD,S] so all WMMA operands
// load as contiguous 16B runs.
// ---------------------------------------------------------------------------
__global__ void __launch_bounds__(128)
k_attn(const _Float16* __restrict__ Q, const _Float16* __restrict__ K,
       const _Float16* __restrict__ Vt, _Float16* __restrict__ ctx) {
  __shared__ _Float16 plds[4][16 * 32];         // 16x32 f16 P tile per wave
  const int wslot = threadIdx.x >> 5;
  const int wid   = blockIdx.x * 4 + wslot;
  const int qt    = wid & 255;                  // SEQ/16 = 256 q tiles
  const int bh    = wid >> 8;                   // 0 .. B*HEADS-1
  const int qbase = qt * 16;

  const size_t head_off = (size_t)bh * SEQ * HD;
  const _Float16* qp = Q  + head_off + (size_t)qbase * HD;
  const _Float16* kp = K  + head_off;
  const _Float16* vp = Vt + head_off;           // [HD][SEQ] for this head

  // Q fragments are kv-loop invariant (d 0-31 and 32-63)
  const v16h aq0 = load_a_frag(qp,      HD);
  const v16h aq1 = load_a_frag(qp + 32, HD);

  v8f o0 = {}, o1 = {}, o2 = {}, o3 = {};
  float mrow[8], lrow[8];
#pragma unroll
  for (int r = 0; r < 8; ++r) { mrow[r] = -3.0e38f; lrow[r] = 0.0f; }

  const int lane = threadIdx.x & 31;
  const int colL = lane & 15;
  const int rhi  = (lane >> 4) * 8;
  const float scale = 0.125f;                   // 1/sqrt(64)
  const int nblk = (qbase + 47) >> 5;           // kv blocks with kb <= qbase

  for (int blk = 0; blk < nblk; ++blk) {
    const int kb = blk * 32;
    const _Float16* kblk = kp + (size_t)kb * HD;

    // scores tile 16x32 = [c0 | c1], K-dim 64 chained as 2x32
    v8f z = {};
    v8f c0 = wmma_f16(aq0, load_b_frag_t(kblk,      HD), z);
    c0     = wmma_f16(aq1, load_b_frag_t(kblk + 32, HD), c0);
    v8f c1 = wmma_f16(aq0, load_b_frag_t(kblk + 16 * HD,      HD), z);
    c1     = wmma_f16(aq1, load_b_frag_t(kblk + 16 * HD + 32, HD), c1);

    float p0[8], p1[8], bm[8];
#pragma unroll
    for (int r = 0; r < 8; ++r) {
      const int row = qbase + r + rhi;
      float s0 = c0[r] * scale;
      float s1 = c1[r] * scale;
      if (kb + colL > row)      s0 = -3.0e38f;  // causal mask
      if (kb + 16 + colL > row) s1 = -3.0e38f;
      p0[r] = s0; p1[r] = s1;
      bm[r] = fmaxf(s0, s1);
    }
    // row max across the 16 lanes of each half-wave
#pragma unroll
    for (int r = 0; r < 8; ++r) {
      float v = bm[r];
      v = fmaxf(v, __shfl_xor(v, 1, 32));
      v = fmaxf(v, __shfl_xor(v, 2, 32));
      v = fmaxf(v, __shfl_xor(v, 4, 32));
      v = fmaxf(v, __shfl_xor(v, 8, 32));
      bm[r] = v;
    }
    float alpha[8];
#pragma unroll
    for (int r = 0; r < 8; ++r) {
      const float mnew = fmaxf(mrow[r], bm[r]);
      alpha[r] = __expf(mrow[r] - mnew);
      mrow[r]  = mnew;
      p0[r] = __expf(p0[r] - mnew);
      p1[r] = __expf(p1[r] - mnew);
      float ps = p0[r] + p1[r];
      ps += __shfl_xor(ps, 1, 32);
      ps += __shfl_xor(ps, 2, 32);
      ps += __shfl_xor(ps, 4, 32);
      ps += __shfl_xor(ps, 8, 32);
      lrow[r] = lrow[r] * alpha[r] + ps;
    }
#pragma unroll
    for (int r = 0; r < 8; ++r) {
      o0[r] *= alpha[r]; o1[r] *= alpha[r];
      o2[r] *= alpha[r]; o3[r] *= alpha[r];
    }

    // stage P (C layout) through LDS, re-read in A-fragment layout
    _Float16* t = plds[wslot];
#pragma unroll
    for (int r = 0; r < 8; ++r) {
      const int row = r + rhi;
      t[row * 32 + colL]      = (_Float16)p0[r];
      t[row * 32 + 16 + colL] = (_Float16)p1[r];
    }
    const v16h ap = load_a_frag(t, 32);         // same-wave LDS RAW: in-order

    // V fragments from transposed storage: row n = head dim d, contiguous kv
    const _Float16* vblk = vp + kb;
    o0 = wmma_f16(ap, load_b_frag_t(vblk,            SEQ), o0);
    o1 = wmma_f16(ap, load_b_frag_t(vblk + 16 * SEQ, SEQ), o1);
    o2 = wmma_f16(ap, load_b_frag_t(vblk + 32 * SEQ, SEQ), o2);
    o3 = wmma_f16(ap, load_b_frag_t(vblk + 48 * SEQ, SEQ), o3);
  }

  // epilogue: normalize, write ctx as [B, S, E] f16 (E = h*64 + d)
  const int b = bh / HEADS;
  const int h = bh - b * HEADS;
#pragma unroll
  for (int r = 0; r < 8; ++r) {
    const float inv = 1.0f / lrow[r];
    const size_t row = (size_t)b * SEQ + qbase + r + rhi;
    _Float16* dst = ctx + row * EMB + (size_t)h * HD + colL;
    dst[0]  = (_Float16)(o0[r] * inv);
    dst[16] = (_Float16)(o1[r] * inv);
    dst[32] = (_Float16)(o2[r] * inv);
    dst[48] = (_Float16)(o3[r] * inv);
  }
}

// ---------------------------------------------------------------------------
extern "C" void kernel_launch(void* const* d_in, const int* in_sizes, int n_in,
                              void* d_out, int out_size, void* d_ws, size_t ws_size,
                              hipStream_t stream) {
  const float* x  = (const float*)d_in[0];
  const float* Wq = (const float*)d_in[1];
  const float* bq = (const float*)d_in[2];
  const float* Wk = (const float*)d_in[3];
  const float* bk = (const float*)d_in[4];
  const float* Wv = (const float*)d_in[5];
  const float* bv = (const float*)d_in[6];
  const float* Wo = (const float*)d_in[7];
  const float* bo = (const float*)d_in[8];

  const size_t NX = (size_t)BATCH * SEQ * EMB;  // 6,291,456
  const size_t NW = (size_t)EMB * EMB;          //   589,824

  _Float16* ws  = (_Float16*)d_ws;
  _Float16* xh  = ws;                 // also reused as ctx after QKV GEMMs
  _Float16* wqt = xh  + NX;
  _Float16* wkt = wqt + NW;
  _Float16* wvt = wkt + NW;
  _Float16* wot = wvt + NW;
  _Float16* qh  = wot + NW;
  _Float16* kh  = qh  + NX;
  _Float16* vth = kh  + NX;           // V transposed: [B,H,HD,SEQ]
  _Float16* ch  = xh;                 // alias: xh dead after projections

  // 1) convert inputs to f16 (weights transposed for contiguous B-fragments)
  k_cvt  <<<(int)((NX + 255) / 256), 256, 0, stream>>>(x, xh, (int)NX);
  k_cvt_t<<<(int)((NW + 255) / 256), 256, 0, stream>>>(Wq, wqt);
  k_cvt_t<<<(int)((NW + 255) / 256), 256, 0, stream>>>(Wk, wkt);
  k_cvt_t<<<(int)((NW + 255) / 256), 256, 0, stream>>>(Wv, wvt);
  k_cvt_t<<<(int)((NW + 255) / 256), 256, 0, stream>>>(Wo, wot);

  // 2) projections: (B*S/32)*(EMB/64) = 3072 waves, 8 waves/block -> 384 blocks
  const int gemm_blocks = (BATCH * SEQ / 32) * (EMB / 64) / 8;
  k_gemm<<<gemm_blocks, 256, 0, stream>>>(xh, wqt, bq, qh,  nullptr, 1);
  k_gemm<<<gemm_blocks, 256, 0, stream>>>(xh, wkt, bk, kh,  nullptr, 1);
  k_gemm<<<gemm_blocks, 256, 0, stream>>>(xh, wvt, bv, vth, nullptr, 2);

  // 3) causal flash attention: B*H*(S/16) = 6144 waves, 4 waves/block
  k_attn<<<(BATCH * HEADS * (SEQ / 16)) / 4, 128, 0, stream>>>(qh, kh, vth, ch);

  // 4) output projection -> fp32 d_out
  k_gemm<<<gemm_blocks, 256, 0, stream>>>(ch, wot, bo, nullptr, (float*)d_out, 0);
}
